// MultiHeadAttention_86870008529338
// MI455X (gfx1250) — matrix operand order repaired and checked by reading, not compiled, not run on hardware
//
#include <hip/hip_runtime.h>

// ---------------------------------------------------------------------------
// Problem constants (reference: B=2, T=2048, C=1024, H=16, DH=64)
// ---------------------------------------------------------------------------
constexpr int B_ = 2, T_ = 2048, C_ = 1024, H_ = 16, DH_ = 64;

// Workspace layout, in _Float16 elements (total 24M halfs = 48 MB)
constexpr long OFF_X  = 0;                          // x in f16        [B*T][C]      4M
constexpr long OFF_W  = 4194304;                    // Wq,Wk,Wv^T f16  [3C][C]       3M
constexpr long OFF_W0 = OFF_W  + 3145728;           // W0 f16          [C][C]        1M
constexpr long OFF_Q  = OFF_W0 + 1048576;           // Q f16           [B,H,T,DH]    4M
constexpr long OFF_K  = OFF_Q  + 4194304;           // K f16           [B,H,T,DH]    4M
constexpr long OFF_VT = OFF_K  + 4194304;           // V^T f16         [B,H,DH,T]    4M
constexpr long OFF_Z  = OFF_VT + 4194304;           // attn out f16    [B,T,C]       4M

typedef _Float16 half8  __attribute__((ext_vector_type(8)));
typedef _Float16 half16 __attribute__((ext_vector_type(16)));
typedef float    v8f    __attribute__((ext_vector_type(8)));

static __device__ inline half16 cat8(half8 lo, half8 hi) {
  return __builtin_shufflevector(lo, hi, 0, 1, 2, 3, 4, 5, 6, 7,
                                         8, 9, 10, 11, 12, 13, 14, 15);
}

static __device__ inline v8f wmma_f16(half16 a, half16 b, v8f c) {
  // D = A(16x32 f16) * B(32x16 f16) + C(16x16 f32)
  return __builtin_amdgcn_wmma_f32_16x16x32_f16(
      /*neg_a=*/false, a, /*neg_b=*/false, b,
      /*c_mod=*/(short)0, c, /*reuse_a=*/false, /*reuse_b=*/false);
}

// A-operand layout (contiguous K): lane<16 -> M=ln, K in {0..15};
//                                  lane>=16 -> M=ln, K in {16..31}
static __device__ inline half16 ldA(const _Float16* row, int k, int aoff) {
  return *(const half16*)(row + k + aoff);
}

// ---------------------------------------------------------------------------
// Kernel 0: f32 -> f16 conversion + weight transpose.
// ---------------------------------------------------------------------------
__global__ __launch_bounds__(256) void convert_kernel(
    const float* __restrict__ x,  const float* __restrict__ Wq,
    const float* __restrict__ Wk, const float* __restrict__ Wv,
    const float* __restrict__ W0, _Float16* __restrict__ ws) {
  const long total = 4194304L + 3145728L + 1048576L;
  for (long i = (long)blockIdx.x * blockDim.x + threadIdx.x; i < total;
       i += (long)gridDim.x * blockDim.x) {
    if (i < 4194304L) {
      ws[OFF_X + i] = (_Float16)x[i];
    } else if (i < 4194304L + 3145728L) {
      long j = i - 4194304L;
      int sel = (int)(j >> 20);          // 0=q 1=k 2=v (1M each)
      int r   = (int)(j & 1048575L);
      int n = r >> 10;                   // output column h*DH+d
      int c = r & 1023;
      int h = n >> 6, d = n & 63;
      const float* W = (sel == 0) ? Wq : (sel == 1) ? Wk : Wv;
      ws[OFF_W + j] = (_Float16)W[((long)h * C_ + c) * DH_ + d];
    } else {
      long j = i - 4194304L - 3145728L;
      ws[OFF_W0 + j] = (_Float16)W0[j];
    }
  }
}

// ---------------------------------------------------------------------------
// Kernel 1: fused QKV projection GEMM, software-pipelined.
//   (B*T x C) @ (C x 3C)  ->  Q,K [B,H,T,DH] f16 ; V stored transposed
//   [B,H,DH,T]. One wave computes a 32(M) x 64(N) tile: 8 accumulators,
//   k-step 32, next-step A/B fragments prefetched into registers.
// ---------------------------------------------------------------------------
__global__ __launch_bounds__(256) void qkv_gemm_kernel(
    _Float16* __restrict__ ws, const float* __restrict__ bq,
    const float* __restrict__ bk, const float* __restrict__ bv) {
  const int lane = threadIdx.x & 31;
  const int wave = threadIdx.x >> 5;
  const int hi = lane >> 4;
  const int ln = lane & 15;
  const int NS = (3 * C_) / 64;        // 48 column strips
  int task = blockIdx.x * 8 + wave;    // 128 * 48 = 6144 tasks, exact
  int m0 = (task / NS) * 32;
  int n0 = (task % NS) * 64;

  const _Float16* A  = ws + OFF_X;     // [M][C]
  const _Float16* Bt = ws + OFF_W;     // [3C][C]  (row = output column)
  const _Float16* arow0 = A + (long)(m0 + ln) * C_;
  const _Float16* arow1 = arow0 + 16L * C_;
  const int aoff = hi * 16;
  const _Float16* bp0 = Bt + (long)(n0 + 0  + ln) * C_ + hi * 16;
  const _Float16* bp1 = Bt + (long)(n0 + 16 + ln) * C_ + hi * 16;
  const _Float16* bp2 = Bt + (long)(n0 + 32 + ln) * C_ + hi * 16;
  const _Float16* bp3 = Bt + (long)(n0 + 48 + ln) * C_ + hi * 16;

  v8f acc[2][4];
#pragma unroll
  for (int g = 0; g < 2; ++g)
#pragma unroll
    for (int t = 0; t < 4; ++t) acc[g][t] = (v8f){0,0,0,0,0,0,0,0};

  // prologue: fragments for k = 0
  half16 a0 = ldA(arow0, 0, aoff), a1 = ldA(arow1, 0, aoff);
  half16 b0 = *(const half16*)bp0, b1 = *(const half16*)bp1;
  half16 b2 = *(const half16*)bp2, b3 = *(const half16*)bp3;

  for (int k = 0; k < C_ - 32; k += 32) {
    const int kn = k + 32;
    // L2 prefetch a few steps ahead
    if (k + 256 < C_) {
      __builtin_prefetch(arow0 + k + 256, 0, 3);
      __builtin_prefetch(bp0 + k + 256, 0, 3);
    }
    // next-step fragments (issued before the WMMAs consume current ones)
    half16 na0 = ldA(arow0, kn, aoff), na1 = ldA(arow1, kn, aoff);
    half16 nb0 = *(const half16*)(bp0 + kn), nb1 = *(const half16*)(bp1 + kn);
    half16 nb2 = *(const half16*)(bp2 + kn), nb3 = *(const half16*)(bp3 + kn);
    acc[0][0] = wmma_f16(a0, b0, acc[0][0]);
    acc[0][1] = wmma_f16(a0, b1, acc[0][1]);
    acc[0][2] = wmma_f16(a0, b2, acc[0][2]);
    acc[0][3] = wmma_f16(a0, b3, acc[0][3]);
    acc[1][0] = wmma_f16(a1, b0, acc[1][0]);
    acc[1][1] = wmma_f16(a1, b1, acc[1][1]);
    acc[1][2] = wmma_f16(a1, b2, acc[1][2]);
    acc[1][3] = wmma_f16(a1, b3, acc[1][3]);
    a0 = na0; a1 = na1; b0 = nb0; b1 = nb1; b2 = nb2; b3 = nb3;
  }
  acc[0][0] = wmma_f16(a0, b0, acc[0][0]);
  acc[0][1] = wmma_f16(a0, b1, acc[0][1]);
  acc[0][2] = wmma_f16(a0, b2, acc[0][2]);
  acc[0][3] = wmma_f16(a0, b3, acc[0][3]);
  acc[1][0] = wmma_f16(a1, b0, acc[1][0]);
  acc[1][1] = wmma_f16(a1, b1, acc[1][1]);
  acc[1][2] = wmma_f16(a1, b2, acc[1][2]);
  acc[1][3] = wmma_f16(a1, b3, acc[1][3]);

  _Float16* Qo = ws + OFF_Q;
  _Float16* Ko = ws + OFF_K;
  _Float16* Vo = ws + OFF_VT;
  const int proj = n0 >> 10;           // uniform within the 64-wide strip
  const int nbase = n0 & 1023;
  const float* bias = (proj == 0) ? bq : (proj == 1) ? bk : bv;
#pragma unroll
  for (int g = 0; g < 2; ++g) {
#pragma unroll
    for (int t = 0; t < 4; ++t) {
      int ncol = nbase + t * 16 + ln;
      int h = ncol >> 6, d = ncol & 63;
      float bsv = bias[ncol];
#pragma unroll
      for (int v = 0; v < 8; ++v) {
        int m = m0 + g * 16 + v + hi * 8;   // C/D layout row
        int b = m >> 11, tt = m & 2047;
        _Float16 hval = (_Float16)(acc[g][t][v] + bsv);
        if (proj == 2)      Vo[(((long)b * H_ + h) * DH_ + d) * T_ + tt] = hval;
        else if (proj == 1) Ko[(((long)b * H_ + h) * T_ + tt) * DH_ + d] = hval;
        else                Qo[(((long)b * H_ + h) * T_ + tt) * DH_ + d] = hval;
      }
    }
  }
}

// ---------------------------------------------------------------------------
// Kernel 2: causal flash attention, pipelined. One wave per (b,h,16-row tile).
// Per 32-key step: 4 score WMMAs, then V loads + next-K prefetch overlap the
// softmax VALU work and the LDS-based C->A transpose of P, then 4 PV WMMAs.
// ---------------------------------------------------------------------------
__global__ __launch_bounds__(256) void attn_kernel(_Float16* __restrict__ ws) {
  __shared__ _Float16 plds[8][16][32];         // 8 KB, one tile per wave
  const int lane = threadIdx.x & 31;
  const int wave = threadIdx.x >> 5;
  const int hi = lane >> 4;
  const int ln = lane & 15;
  const int QT = T_ / 16;                      // 128 Q tiles per (b,h)
  int task = blockIdx.x * 8 + wave;            // 2*16*128 = 4096 tasks, exact
  int bh = task / QT;
  int q0 = (task % QT) * 16;

  const _Float16* Qp = ws + OFF_Q  + (long)bh * T_ * DH_;
  const _Float16* Kp = ws + OFF_K  + (long)bh * T_ * DH_;
  const _Float16* Vp = ws + OFF_VT + (long)bh * DH_ * T_;   // [DH][T]

  // Q as two 16x32 A-operands (K-dim = d)
  half16 qa0, qa1;
  {
    const _Float16* qrow = Qp + (long)(q0 + ln) * DH_;
    qa0 = *(const half16*)(qrow + hi * 16);
    qa1 = *(const half16*)(qrow + 32 + hi * 16);
  }

  v8f oacc[4];
#pragma unroll
  for (int t = 0; t < 4; ++t) oacc[t] = (v8f){0,0,0,0,0,0,0,0};
  float mrow[8], lrow[8];
#pragma unroll
  for (int v = 0; v < 8; ++v) { mrow[v] = -3.0e38f; lrow[v] = 0.0f; }

  const int s_end = (q0 + 16 + 31) & ~31;      // include diagonal tile

  // prologue: K B-operands for s0 = 0  (kb[j][c]: key sub-tile j, d-chunk c)
  half16 kb[2][2];
#pragma unroll
  for (int j = 0; j < 2; ++j) {
    const _Float16* krow = Kp + (long)(j * 16 + ln) * DH_;
    kb[j][0] = *(const half16*)(krow + hi * 16);
    kb[j][1] = *(const half16*)(krow + 32 + hi * 16);
  }

  for (int s0 = 0; s0 < s_end; s0 += 32) {
    // ---- S = Q @ K^T (16 x 32), two 16x16 f32 tiles ----
    v8f sacc[2] = {{0,0,0,0,0,0,0,0},{0,0,0,0,0,0,0,0}};
    sacc[0] = wmma_f16(qa0, kb[0][0], sacc[0]);
    sacc[0] = wmma_f16(qa1, kb[0][1], sacc[0]);
    sacc[1] = wmma_f16(qa0, kb[1][0], sacc[1]);
    sacc[1] = wmma_f16(qa1, kb[1][1], sacc[1]);

    // ---- V B-operands for this tile: overlap with score WMMAs ----
    half16 vb[4];
#pragma unroll
    for (int t = 0; t < 4; ++t)
      vb[t] = *(const half16*)(Vp + (long)(t * 16 + ln) * T_ + s0 + hi * 16);

    // ---- prefetch next K tile (clamped, uniform): overlaps softmax ----
    const int sn = (s0 + 32 < s_end) ? s0 + 32 : s0;
    half16 nkb[2][2];
#pragma unroll
    for (int j = 0; j < 2; ++j) {
      const _Float16* krow = Kp + (long)(sn + j * 16 + ln) * DH_;
      nkb[j][0] = *(const half16*)(krow + hi * 16);
      nkb[j][1] = *(const half16*)(krow + 32 + hi * 16);
    }

    // ---- online softmax (rows live on the 16 lanes of each half) ----
    const bool diag = (s0 + 31 > q0);
    float p0[8], p1[8];
#pragma unroll
    for (int v = 0; v < 8; ++v) {
      int r = q0 + v + hi * 8;                 // global query row
      float x0 = sacc[0][v] * 0.125f;          // 1/sqrt(64)
      float x1 = sacc[1][v] * 0.125f;
      if (diag) {
        if (s0 + ln > r)      x0 = -3.0e38f;
        if (s0 + 16 + ln > r) x1 = -3.0e38f;
      }
      float mx = fmaxf(x0, x1);
#pragma unroll
      for (int msk = 1; msk < 16; msk <<= 1)
        mx = fmaxf(mx, __shfl_xor(mx, msk, 32));
      float newm = fmaxf(mrow[v], mx);
      float alpha = __expf(mrow[v] - newm);
      float e0 = __expf(x0 - newm);
      float e1 = __expf(x1 - newm);
      float rs = e0 + e1;
#pragma unroll
      for (int msk = 1; msk < 16; msk <<= 1)
        rs += __shfl_xor(rs, msk, 32);
      lrow[v] = lrow[v] * alpha + rs;
      mrow[v] = newm;
#pragma unroll
      for (int t = 0; t < 4; ++t) oacc[t][v] *= alpha;
      p0[v] = e0; p1[v] = e1;
    }

    // ---- transpose P: C-layout -> 16x32 A-operand via LDS ----
#pragma unroll
    for (int v = 0; v < 8; ++v) {
      plds[wave][v + hi * 8][ln]      = (_Float16)p0[v];
      plds[wave][v + hi * 8][16 + ln] = (_Float16)p1[v];
    }
    asm volatile("s_wait_dscnt 0x0" ::: "memory");   // intra-wave DS ordering
    half16 pa;
    {
      const _Float16* prow = &plds[wave][ln][0];
      pa = *(const half16*)(prow + hi * 16);
    }

    // ---- O += P @ V (K-dim = 32 keys, N = d) ----
    oacc[0] = wmma_f16(pa, vb[0], oacc[0]);
    oacc[1] = wmma_f16(pa, vb[1], oacc[1]);
    oacc[2] = wmma_f16(pa, vb[2], oacc[2]);
    oacc[3] = wmma_f16(pa, vb[3], oacc[3]);
    asm volatile("s_wait_dscnt 0x0" ::: "memory");   // LDS WAR before next tile

#pragma unroll
    for (int j = 0; j < 2; ++j) {
      kb[j][0] = nkb[j][0];
      kb[j][1] = nkb[j][1];
    }
  }

  // ---- epilogue: O / l, write Z f16 [B,T,C] with heads concatenated ----
  int b = bh / H_, h = bh % H_;
#pragma unroll
  for (int v = 0; v < 8; ++v) {
    float inv = 1.0f / lrow[v];
    int m = q0 + v + hi * 8;
    long base = (long)(b * T_ + m) * C_ + h * DH_;
#pragma unroll
    for (int t = 0; t < 4; ++t)
      ws[OFF_Z + base + t * 16 + ln] = (_Float16)(oacc[t][v] * inv);
  }
}

// ---------------------------------------------------------------------------
// Kernel 3: output projection  out = Z @ W0^T + b0, software-pipelined,
// 32(M) x 64(N) per wave, f32 result straight to d_out.
// ---------------------------------------------------------------------------
__global__ __launch_bounds__(256) void out_gemm_kernel(
    const _Float16* __restrict__ ws, const float* __restrict__ b0v,
    float* __restrict__ out) {
  const int lane = threadIdx.x & 31;
  const int wave = threadIdx.x >> 5;
  const int hi = lane >> 4;
  const int ln = lane & 15;
  const int NS = C_ / 64;              // 16
  int task = blockIdx.x * 8 + wave;    // 128 * 16 = 2048 tasks, exact
  int m0 = (task / NS) * 32;
  int n0 = (task % NS) * 64;

  const _Float16* A  = ws + OFF_Z;     // [M][C]
  const _Float16* Bt = ws + OFF_W0;    // [C][C] (row = output feature)
  const _Float16* arow0 = A + (long)(m0 + ln) * C_;
  const _Float16* arow1 = arow0 + 16L * C_;
  const int aoff = hi * 16;
  const _Float16* bp0 = Bt + (long)(n0 + 0  + ln) * C_ + hi * 16;
  const _Float16* bp1 = Bt + (long)(n0 + 16 + ln) * C_ + hi * 16;
  const _Float16* bp2 = Bt + (long)(n0 + 32 + ln) * C_ + hi * 16;
  const _Float16* bp3 = Bt + (long)(n0 + 48 + ln) * C_ + hi * 16;

  v8f acc[2][4];
#pragma unroll
  for (int g = 0; g < 2; ++g)
#pragma unroll
    for (int t = 0; t < 4; ++t) acc[g][t] = (v8f){0,0,0,0,0,0,0,0};

  half16 a0 = ldA(arow0, 0, aoff), a1 = ldA(arow1, 0, aoff);
  half16 b0 = *(const half16*)bp0, b1 = *(const half16*)bp1;
  half16 b2 = *(const half16*)bp2, b3 = *(const half16*)bp3;

  for (int k = 0; k < C_ - 32; k += 32) {
    const int kn = k + 32;
    if (k + 256 < C_) {
      __builtin_prefetch(arow0 + k + 256, 0, 3);
      __builtin_prefetch(bp0 + k + 256, 0, 3);
    }
    half16 na0 = ldA(arow0, kn, aoff), na1 = ldA(arow1, kn, aoff);
    half16 nb0 = *(const half16*)(bp0 + kn), nb1 = *(const half16*)(bp1 + kn);
    half16 nb2 = *(const half16*)(bp2 + kn), nb3 = *(const half16*)(bp3 + kn);
    acc[0][0] = wmma_f16(a0, b0, acc[0][0]);
    acc[0][1] = wmma_f16(a0, b1, acc[0][1]);
    acc[0][2] = wmma_f16(a0, b2, acc[0][2]);
    acc[0][3] = wmma_f16(a0, b3, acc[0][3]);
    acc[1][0] = wmma_f16(a1, b0, acc[1][0]);
    acc[1][1] = wmma_f16(a1, b1, acc[1][1]);
    acc[1][2] = wmma_f16(a1, b2, acc[1][2]);
    acc[1][3] = wmma_f16(a1, b3, acc[1][3]);
    a0 = na0; a1 = na1; b0 = nb0; b1 = nb1; b2 = nb2; b3 = nb3;
  }
  acc[0][0] = wmma_f16(a0, b0, acc[0][0]);
  acc[0][1] = wmma_f16(a0, b1, acc[0][1]);
  acc[0][2] = wmma_f16(a0, b2, acc[0][2]);
  acc[0][3] = wmma_f16(a0, b3, acc[0][3]);
  acc[1][0] = wmma_f16(a1, b0, acc[1][0]);
  acc[1][1] = wmma_f16(a1, b1, acc[1][1]);
  acc[1][2] = wmma_f16(a1, b2, acc[1][2]);
  acc[1][3] = wmma_f16(a1, b3, acc[1][3]);

#pragma unroll
  for (int g = 0; g < 2; ++g) {
#pragma unroll
    for (int t = 0; t < 4; ++t) {
      int ncol = n0 + t * 16 + ln;
      float bias = b0v[ncol];
#pragma unroll
      for (int v = 0; v < 8; ++v) {
        int m = m0 + g * 16 + v + hi * 8;
        out[(long)m * C_ + ncol] = acc[g][t][v] + bias;
      }
    }
  }
}

// ---------------------------------------------------------------------------
// Host entry
// ---------------------------------------------------------------------------
extern "C" void kernel_launch(void* const* d_in, const int* in_sizes, int n_in,
                              void* d_out, int out_size, void* d_ws,
                              size_t ws_size, hipStream_t stream) {
  (void)in_sizes; (void)n_in; (void)out_size; (void)ws_size;
  const float* x  = (const float*)d_in[0];
  const float* Wq = (const float*)d_in[1];
  const float* bq = (const float*)d_in[2];
  const float* Wk = (const float*)d_in[3];
  const float* bk = (const float*)d_in[4];
  const float* Wv = (const float*)d_in[5];
  const float* bv = (const float*)d_in[6];
  const float* W0 = (const float*)d_in[7];
  const float* b0 = (const float*)d_in[8];
  _Float16* ws = (_Float16*)d_ws;          // needs 48 MB of scratch
  float* out = (float*)d_out;

  convert_kernel <<<8192, 256, 0, stream>>>(x, Wq, Wk, Wv, W0, ws);
  qkv_gemm_kernel<<<768, 256, 0, stream>>>(ws, bq, bk, bv);
  attn_kernel    <<<512, 256, 0, stream>>>(ws);
  out_gemm_kernel<<<256, 256, 0, stream>>>(ws, b0, out);
}
